// CausalSelfAttention_6133213298769
// MI455X (gfx1250) — matrix operand order repaired and checked by reading, not compiled, not run on hardware
//
#include <hip/hip_runtime.h>
#include <hip/hip_bf16.h>

typedef __bf16 bf16;
typedef __attribute__((ext_vector_type(16))) __bf16 v16bf;
typedef __attribute__((ext_vector_type(8)))  float  v8f;

#define S_LEN 4096
#define C_DIM 1024
#define NHEAD 16
#define HDIM  64

union U4 { uint4 u; __bf16 h[8]; };

// Load 8 contiguous bf16 (16B aligned) into elements [e0, e0+8) of a v16bf.
static __device__ __forceinline__ void ld8(v16bf& v, int e0, const bf16* p) {
    U4 t; t.u = *reinterpret_cast<const uint4*>(p);
#pragma unroll
    for (int i = 0; i < 8; ++i) v[e0 + i] = t.h[i];
}
// Load a full B operand (two 8-bf16 chunks at p and p+16 per lane).
static __device__ __forceinline__ void ldB(v16bf& v, const bf16* p) {
    ld8(v, 0, p); ld8(v, 8, p + 16);
}
// Load an A operand pair (rows m and m+16) for K-slice at offset k.
static __device__ __forceinline__ void ldA(v16bf& v0, v16bf& v1,
                                           const bf16* ar0, const bf16* ar1,
                                           int k, int half) {
    ld8(v0, 0, ar0 + k + half * 8); ld8(v0, 8, ar0 + k + 16 + half * 8);
    ld8(v1, 0, ar1 + k + half * 8); ld8(v1, 8, ar1 + k + 16 + half * 8);
}

static __device__ __forceinline__ float rmax16(float v) {
#pragma unroll
    for (int m = 1; m < 16; m <<= 1) v = fmaxf(v, __shfl_xor(v, m, 32));
    return v;
}
static __device__ __forceinline__ float rsum16(float v) {
#pragma unroll
    for (int m = 1; m < 16; m <<= 1) v += __shfl_xor(v, m, 32);
    return v;
}

// ---------------------------------------------------------------------------
// f32 -> bf16 conversion, 4 elements / thread (b128 load, b64 store)
// ---------------------------------------------------------------------------
__global__ void cvt_f32_bf16(const float* __restrict__ in, bf16* __restrict__ out, int n) {
    int i = (blockIdx.x * 256 + threadIdx.x) * 4;
    if (i < n) {
        float4 f = *reinterpret_cast<const float4*>(in + i);
        union { uint2 u; bf16 h[4]; } t;
        t.h[0] = (bf16)f.x; t.h[1] = (bf16)f.y; t.h[2] = (bf16)f.z; t.h[3] = (bf16)f.w;
        *reinterpret_cast<uint2*>(out + i) = t.u;
    }
}

// ---------------------------------------------------------------------------
// GEMM mainloop: one wave computes a 32(M) x 64(N) f32 tile of
// A(MxK, bf16 row-major) * B(NxK, bf16 row-major)^T, K multiple of 64.
// A tiles double-buffered across K-steps; B tiles ping-pong (bP/bQ) so the
// load for WMMA-pair nt+1 issues while pair nt computes.
// Live VGPRs ~130 (acc 64 + A 32 + B 16 + addressing) -> no spills.
// ---------------------------------------------------------------------------
static __device__ __forceinline__ void gemm32x64(
    const bf16* __restrict__ A, const bf16* __restrict__ B,
    int mtile, int ntile, int K, int lane, int half, v8f acc[8]) {

    const bf16* ar0 = A + (size_t)(mtile * 32 + lane) * K;
    const bf16* ar1 = ar0 + (size_t)16 * K;
    const bf16* br[4];
#pragma unroll
    for (int nt = 0; nt < 4; ++nt)
        br[nt] = B + (size_t)(ntile * 64 + nt * 16 + lane) * K + half * 8;

    v16bf aX0, aX1, aY0, aY1, bP, bQ;

#define WMMA_PAIR(A0, A1, BB, NT)                                              \
    acc[NT]     = __builtin_amdgcn_wmma_f32_16x16x32_bf16(                     \
        false, A0, false, BB, (short)0, acc[NT], false, false);                \
    acc[4 + NT] = __builtin_amdgcn_wmma_f32_16x16x32_bf16(                     \
        false, A1, false, BB, (short)0, acc[4 + NT], false, false)

    // prologue
    ldA(aX0, aX1, ar0, ar1, 0, half);
    ldB(bP, br[0]);

    for (int kb = 0; kb < K; kb += 64) {
        // ---- half-step 1: compute with aX (kb), prefetch aY (kb+32)
        ldA(aY0, aY1, ar0, ar1, kb + 32, half);
        ldB(bQ, br[1] + kb);
        WMMA_PAIR(aX0, aX1, bP, 0);
        ldB(bP, br[2] + kb);
        WMMA_PAIR(aX0, aX1, bQ, 1);
        ldB(bQ, br[3] + kb);
        WMMA_PAIR(aX0, aX1, bP, 2);
        ldB(bP, br[0] + kb + 32);
        WMMA_PAIR(aX0, aX1, bQ, 3);

        // ---- half-step 2: compute with aY (kb+32), prefetch aX (kb+64)
        if (kb + 64 < K) ldA(aX0, aX1, ar0, ar1, kb + 64, half);
        ldB(bQ, br[1] + kb + 32);
        WMMA_PAIR(aY0, aY1, bP, 0);
        ldB(bP, br[2] + kb + 32);
        WMMA_PAIR(aY0, aY1, bQ, 1);
        ldB(bQ, br[3] + kb + 32);
        WMMA_PAIR(aY0, aY1, bP, 2);
        if (kb + 64 < K) ldB(bP, br[0] + kb + 64);
        WMMA_PAIR(aY0, aY1, bQ, 3);
    }
#undef WMMA_PAIR
}

// ---------------------------------------------------------------------------
// QKV GEMM (4096 x 3072 x 1024) with scatter into Qh[h][s][d], Kh[h][s][d],
// Vt[h][d][s]. grid = (48, 32), block = 128 (4 waves, each a 32x64 tile).
// ---------------------------------------------------------------------------
__global__ void __launch_bounds__(128, 1)
qkv_gemm(const bf16* __restrict__ X, const bf16* __restrict__ W,
         const float* __restrict__ bias,
         bf16* __restrict__ Qh, bf16* __restrict__ Kh, bf16* __restrict__ Vt) {
    const int wave  = threadIdx.x >> 5;
    const int mtile = blockIdx.y * 4 + wave;
    const int ntile = blockIdx.x;
    const int l     = threadIdx.x & 31;
    const int lane  = l & 15;
    const int half  = l >> 4;

    v8f acc[8] = {};
    gemm32x64(X, W, mtile, ntile, C_DIM, lane, half, acc);

#pragma unroll
    for (int nt = 0; nt < 4; ++nt) {
        const int o = ntile * 64 + nt * 16 + lane;
        const float bo = bias[o];
#pragma unroll
        for (int mt = 0; mt < 2; ++mt) {
#pragma unroll
            for (int r = 0; r < 8; ++r) {
                const int srow = mtile * 32 + mt * 16 + r + half * 8;
                const float v = acc[mt * 4 + nt][r] + bo;
                if (o < C_DIM) {
                    int h = o >> 6, d = o & 63;
                    Qh[((size_t)h * S_LEN + srow) * HDIM + d] = (bf16)v;
                } else if (o < 2 * C_DIM) {
                    int oo = o - C_DIM, h = oo >> 6, d = oo & 63;
                    Kh[((size_t)h * S_LEN + srow) * HDIM + d] = (bf16)v;
                } else {
                    int oo = o - 2 * C_DIM, h = oo >> 6, d = oo & 63;
                    Vt[((size_t)h * HDIM + d) * S_LEN + srow] = (bf16)v;
                }
            }
        }
    }
}

// ---------------------------------------------------------------------------
// Flash attention: one wave per (32-query tile, head); 32-key blocks with
// online softmax and causal block skip. grid = (S/32=128, NH=16), block = 32.
// K tile and V tile loads are shared across the two 16-query sub-tiles.
// ---------------------------------------------------------------------------
__global__ void __launch_bounds__(32, 1)
attn_flash(const bf16* __restrict__ Qh, const bf16* __restrict__ Kh,
           const bf16* __restrict__ Vt, const unsigned char* __restrict__ pmask,
           bf16* __restrict__ AO) {
    __shared__ float lds[32][36];

    const int qtile = blockIdx.x;
    const int h     = blockIdx.y;
    const int l     = threadIdx.x & 31;
    const int lane  = l & 15;
    const int half  = l >> 4;
    const int qbase = qtile * 32;
    const float scale = 0.125f;  // 1/sqrt(64)

    // Q tiles: 2 sub-tiles x 2 K-slices (d 0..31 / 32..63)
    v16bf aq[2][2];
#pragma unroll
    for (int mt = 0; mt < 2; ++mt) {
        const bf16* qrow = Qh + ((size_t)h * S_LEN + qbase + mt * 16 + lane) * HDIM;
        ld8(aq[mt][0], 0, qrow + half * 8);
        ld8(aq[mt][0], 8, qrow + 16 + half * 8);
        ld8(aq[mt][1], 0, qrow + 32 + half * 8);
        ld8(aq[mt][1], 8, qrow + 48 + half * 8);
    }

    float mrow[2][8], lrow[2][8];
#pragma unroll
    for (int mt = 0; mt < 2; ++mt)
#pragma unroll
        for (int r = 0; r < 8; ++r) { mrow[mt][r] = -INFINITY; lrow[mt][r] = 0.0f; }
    v8f acc[2][4] = {};

    const int kend = qbase + 32;  // causal: keys <= max query index in tile

    for (int kb = 0; kb < kend; kb += 32) {
        // --- scores: 2 key sub-tiles x 2 query sub-tiles
        v8f sc[2][2];
#pragma unroll
        for (int t = 0; t < 2; ++t) {
            const int key = kb + t * 16 + lane;
            const bf16* krow = Kh + ((size_t)h * S_LEN + key) * HDIM;
            v16bf b0, b1;
            ldB(b0, krow + half * 8);
            ldB(b1, krow + 32 + half * 8);
#pragma unroll
            for (int mt = 0; mt < 2; ++mt) {
                v8f s = {};
                s = __builtin_amdgcn_wmma_f32_16x16x32_bf16(false, aq[mt][0], false, b0, (short)0, s, false, false);
                s = __builtin_amdgcn_wmma_f32_16x16x32_bf16(false, aq[mt][1], false, b1, (short)0, s, false, false);
                sc[mt][t] = s;
            }
        }

        // --- scale + causal/padding mask
#pragma unroll
        for (int t = 0; t < 2; ++t) {
            const int key = kb + t * 16 + lane;
            const bool pmk = pmask[key] != 0;
#pragma unroll
            for (int mt = 0; mt < 2; ++mt) {
#pragma unroll
                for (int r = 0; r < 8; ++r) {
                    const int qi = qbase + mt * 16 + r + half * 8;
                    float v = sc[mt][t][r] * scale;
                    if (key > qi || pmk) v = -INFINITY;
                    sc[mt][t][r] = v;
                }
            }
        }

        // --- online softmax update + store P to LDS
#pragma unroll
        for (int mt = 0; mt < 2; ++mt) {
            float alpha[8];
#pragma unroll
            for (int r = 0; r < 8; ++r) {
                float tm = rmax16(fmaxf(sc[mt][0][r], sc[mt][1][r]));
                float mn = fmaxf(mrow[mt][r], tm);
                alpha[r] = __expf(mrow[mt][r] - mn);
                mrow[mt][r] = mn;
                float p0 = __expf(sc[mt][0][r] - mn);
                float p1 = __expf(sc[mt][1][r] - mn);
                sc[mt][0][r] = p0;
                sc[mt][1][r] = p1;
                lrow[mt][r] = lrow[mt][r] * alpha[r] + rsum16(p0 + p1);
            }
#pragma unroll
            for (int nt = 0; nt < 4; ++nt)
#pragma unroll
                for (int r = 0; r < 8; ++r) acc[mt][nt][r] *= alpha[r];
#pragma unroll
            for (int t = 0; t < 2; ++t)
#pragma unroll
                for (int r = 0; r < 8; ++r)
                    lds[mt * 16 + r + half * 8][t * 16 + lane] = sc[mt][t][r];
        }
        __syncthreads();

        // --- P in A-layout (16x32 per query sub-tile)
        v16bf ap[2];
#pragma unroll
        for (int mt = 0; mt < 2; ++mt) {
#pragma unroll
            for (int i = 0; i < 8; ++i) {
                ap[mt][i]     = (bf16)lds[mt * 16 + lane][half * 8 + i];
                ap[mt][8 + i] = (bf16)lds[mt * 16 + lane][16 + half * 8 + i];
            }
        }

        // --- P @ V: V tile shared by both query sub-tiles
#pragma unroll
        for (int nt = 0; nt < 4; ++nt) {
            const bf16* vrow = Vt + ((size_t)h * HDIM + nt * 16 + lane) * S_LEN + kb + half * 8;
            v16bf bv;
            ldB(bv, vrow);
#pragma unroll
            for (int mt = 0; mt < 2; ++mt)
                acc[mt][nt] = __builtin_amdgcn_wmma_f32_16x16x32_bf16(
                    false, ap[mt], false, bv, (short)0, acc[mt][nt], false, false);
        }
        __syncthreads();
    }

    // --- finalize: out[s][h*64+d] bf16
#pragma unroll
    for (int mt = 0; mt < 2; ++mt) {
#pragma unroll
        for (int nt = 0; nt < 4; ++nt) {
#pragma unroll
            for (int r = 0; r < 8; ++r) {
                const int srow = qbase + mt * 16 + r + half * 8;
                const int col  = h * HDIM + nt * 16 + lane;
                AO[(size_t)srow * C_DIM + col] = (bf16)(acc[mt][nt][r] / lrow[mt][r]);
            }
        }
    }
}

// ---------------------------------------------------------------------------
// Output projection (4096 x 1024 x 1024) -> f32. grid = (16, 32), block = 128.
// ---------------------------------------------------------------------------
__global__ void __launch_bounds__(128, 1)
oproj_gemm(const bf16* __restrict__ A, const bf16* __restrict__ W,
           const float* __restrict__ bias, float* __restrict__ out) {
    const int wave  = threadIdx.x >> 5;
    const int mtile = blockIdx.y * 4 + wave;
    const int ntile = blockIdx.x;
    const int l     = threadIdx.x & 31;
    const int lane  = l & 15;
    const int half  = l >> 4;

    v8f acc[8] = {};
    gemm32x64(A, W, mtile, ntile, C_DIM, lane, half, acc);

#pragma unroll
    for (int nt = 0; nt < 4; ++nt) {
        const int o = ntile * 64 + nt * 16 + lane;
        const float bo = bias[o];
#pragma unroll
        for (int mt = 0; mt < 2; ++mt) {
#pragma unroll
            for (int r = 0; r < 8; ++r) {
                const int srow = mtile * 32 + mt * 16 + r + half * 8;
                out[(size_t)srow * C_DIM + o] = acc[mt * 4 + nt][r] + bo;
            }
        }
    }
}

extern "C" void kernel_launch(void* const* d_in, const int* in_sizes, int n_in,
                              void* d_out, int out_size, void* d_ws, size_t ws_size,
                              hipStream_t stream) {
    const float* x      = (const float*)d_in[0];
    const unsigned char* mask = (const unsigned char*)d_in[1];
    const float* Wqkv_w = (const float*)d_in[2];
    const float* Wqkv_b = (const float*)d_in[3];
    const float* Wo_w   = (const float*)d_in[4];
    const float* Wo_b   = (const float*)d_in[5];
    float* out = (float*)d_out;

    // Workspace layout (bf16 buffers), 256B aligned
    char* ws = (char*)d_ws;
    size_t off = 0;
    auto carve = [&](size_t bytes) {
        char* p = ws + off;
        off = (off + bytes + 255) & ~(size_t)255;
        return p;
    };
    bf16* xb    = (bf16*)carve((size_t)S_LEN * C_DIM * 2);        // 8 MB
    bf16* wqkvb = (bf16*)carve((size_t)3 * C_DIM * C_DIM * 2);    // 6 MB
    bf16* wob   = (bf16*)carve((size_t)C_DIM * C_DIM * 2);        // 2 MB
    bf16* Qh    = (bf16*)carve((size_t)NHEAD * S_LEN * HDIM * 2); // 8 MB
    bf16* Kh    = (bf16*)carve((size_t)NHEAD * S_LEN * HDIM * 2); // 8 MB
    bf16* Vt    = (bf16*)carve((size_t)NHEAD * S_LEN * HDIM * 2); // 8 MB
    bf16* AO    = (bf16*)carve((size_t)S_LEN * C_DIM * 2);        // 8 MB

    // 1) f32 -> bf16 conversions (4 elems/thread)
    {
        int n = S_LEN * C_DIM;
        cvt_f32_bf16<<<(n / 4 + 255) / 256, 256, 0, stream>>>(x, xb, n);
        n = 3 * C_DIM * C_DIM;
        cvt_f32_bf16<<<(n / 4 + 255) / 256, 256, 0, stream>>>(Wqkv_w, wqkvb, n);
        n = C_DIM * C_DIM;
        cvt_f32_bf16<<<(n / 4 + 255) / 256, 256, 0, stream>>>(Wo_w, wob, n);
    }

    // 2) QKV projection
    qkv_gemm<<<dim3(3 * C_DIM / 64, S_LEN / 128), 128, 0, stream>>>(
        xb, wqkvb, Wqkv_b, Qh, Kh, Vt);

    // 3) Flash attention
    attn_flash<<<dim3(S_LEN / 32, NHEAD), 32, 0, stream>>>(Qh, Kh, Vt, mask, AO);

    // 4) Output projection -> f32 out
    oproj_gemm<<<dim3(C_DIM / 64, S_LEN / 128), 128, 0, stream>>>(AO, wob, Wo_b, out);

    (void)in_sizes; (void)n_in; (void)out_size; (void)ws_size;
}
